// DeformableAttention_34325378629771
// MI455X (gfx1250) — matrix operand order repaired and checked by reading, not compiled, not run on hardware
//
#include <hip/hip_runtime.h>
#include <hip/hip_bf16.h>
#include <math.h>

// B=8, N=2048, M=2048, C=768, H=12, P=4, Dh=64
#define Bb 8
#define Nn 2048
#define Mm 2048
#define Cc 768
#define Hh 12
#define Pp 4
#define Dd 64

typedef __attribute__((ext_vector_type(16))) __bf16 v16bf;
typedef __attribute__((ext_vector_type(8)))  float  v8f;
typedef __attribute__((ext_vector_type(4)))  unsigned v4u;
typedef __attribute__((ext_vector_type(8)))  int    v8i_;
typedef __attribute__((ext_vector_type(4)))  int    v4i_;

__device__ __forceinline__ __bf16 f2bf(float f) {
  unsigned u = __builtin_bit_cast(unsigned, f);
  u = (u + 0x7FFFu + ((u >> 16) & 1u)) >> 16;   // round-to-nearest-even
  unsigned short s = (unsigned short)u;
  return __builtin_bit_cast(__bf16, s);
}

// ---------------------------------------------------------------------------
// Pre-shuffle W[768x768] (f32) into WMMA-B fragment order (bf16):
//   WF[t(24)][nb(48)][lane(32)][h(16)],  col = nb*16 + (lane&15),
//   k = t*32 + ((h>>3)<<4) + ((lane>>4)<<3) + (h&7)  (16-bit B 32x16 ISA layout)
// ---------------------------------------------------------------------------
__global__ __launch_bounds__(256)
void prep_w_frag_kernel(const float* __restrict__ W, __bf16* __restrict__ WF) {
  int idx = blockIdx.x * 256 + threadIdx.x;       // [0, 589824)
  int h    = idx & 15;
  int lane = (idx >> 4) & 31;
  int u    = idx >> 9;
  int nb   = u % 48;
  int t    = u / 48;
  int col  = nb * 16 + (lane & 15);
  int k    = t * 32 + ((h >> 3) << 4) + ((lane >> 4) << 3) + (h & 7);
  WF[idx] = f2bf(W[(size_t)k * Cc + col]);
}

// ---------------------------------------------------------------------------
// Pre-shuffle activation A[16384x768] (f32) into WMMA-A fragment order (bf16):
//   AF[mb(1024)][t(24)][lane(32)][h(16)], row = mb*16 + (lane&15),
//   k = t*32 + ((h>>3)<<4) + ((lane>>4)<<3) + (h&7)  (16-bit A 16x32 layout)
// ---------------------------------------------------------------------------
__global__ __launch_bounds__(256)
void prep_a_frag_kernel(const float* __restrict__ A, __bf16* __restrict__ AF) {
  int idx = blockIdx.x * 256 + threadIdx.x;       // [0, 12582912)
  int h    = idx & 15;
  int lane = (idx >> 4) & 31;
  int u    = idx >> 9;
  int t    = u % 24;
  int mb   = u / 24;
  int row  = mb * 16 + (lane & 15);
  int k    = t * 32 + ((h >> 3) << 4) + ((lane >> 4) << 3) + (h & 7);
  AF[idx] = f2bf(A[(size_t)row * Cc + k]);
}

// ---------------------------------------------------------------------------
// LDS-free, barrier-free WMMA GEMM on pre-fragmented operands.
// Out = A[16384x768] * W[768x768]  (f32 accum).
// PERM==0: natural [row, col]; PERM==1: kp [B,H,M,Dh] layout.
// ---------------------------------------------------------------------------
template <int PERM>
__global__ __launch_bounds__(256)
void gemm_frag_kernel(const __bf16* __restrict__ AF, const __bf16* __restrict__ WF,
                      float* __restrict__ Out) {
  const int lane  = threadIdx.x & 31;
  const int wave  = threadIdx.x >> 5;
  const int mBase = blockIdx.x * 128;
  const int nBase = blockIdx.y * 128;
  const int waveM = (wave & 3) * 32;
  const int waveN = (wave >> 2) * 64;

  const int mb0 = (mBase + waveM) >> 4;   // two A bands: mb0, mb0+1
  const int nb0 = (nBase + waveN) >> 4;   // four B bands: nb0..nb0+3

  const __bf16* a0p = AF + (((size_t)mb0 * 24) * 32 + lane) * 16;
  const __bf16* a1p = AF + (((size_t)(mb0 + 1) * 24) * 32 + lane) * 16;
  const __bf16* bp  = WF + ((size_t)nb0 * 32 + lane) * 16;

  v8f acc[2][4];
#pragma unroll
  for (int i = 0; i < 2; i++)
#pragma unroll
    for (int j = 0; j < 4; j++)
      acc[i][j] = (v8f){0.f, 0.f, 0.f, 0.f, 0.f, 0.f, 0.f, 0.f};

#pragma unroll 2
  for (int t = 0; t < 24; t++) {
    v16bf a0 = *(const v16bf*)(a0p + (size_t)t * 512);
    v16bf a1 = *(const v16bf*)(a1p + (size_t)t * 512);
    const __bf16* bt = bp + (size_t)t * (48 * 512);
    v16bf b0 = *(const v16bf*)(bt);
    v16bf b1 = *(const v16bf*)(bt + 512);
    v16bf b2 = *(const v16bf*)(bt + 1024);
    v16bf b3 = *(const v16bf*)(bt + 1536);

    acc[0][0] = __builtin_amdgcn_wmma_f32_16x16x32_bf16(false, a0, false, b0, (short)0, acc[0][0], false, false);
    acc[0][1] = __builtin_amdgcn_wmma_f32_16x16x32_bf16(false, a0, false, b1, (short)0, acc[0][1], false, false);
    acc[0][2] = __builtin_amdgcn_wmma_f32_16x16x32_bf16(false, a0, false, b2, (short)0, acc[0][2], false, false);
    acc[0][3] = __builtin_amdgcn_wmma_f32_16x16x32_bf16(false, a0, false, b3, (short)0, acc[0][3], false, false);
    acc[1][0] = __builtin_amdgcn_wmma_f32_16x16x32_bf16(false, a1, false, b0, (short)0, acc[1][0], false, false);
    acc[1][1] = __builtin_amdgcn_wmma_f32_16x16x32_bf16(false, a1, false, b1, (short)0, acc[1][1], false, false);
    acc[1][2] = __builtin_amdgcn_wmma_f32_16x16x32_bf16(false, a1, false, b2, (short)0, acc[1][2], false, false);
    acc[1][3] = __builtin_amdgcn_wmma_f32_16x16x32_bf16(false, a1, false, b3, (short)0, acc[1][3], false, false);
  }

  // f32 C/D layout: VGPR r -> M = r + 8*(lane>=16), N = lane&15
#pragma unroll
  for (int mi = 0; mi < 2; mi++)
#pragma unroll
    for (int ni = 0; ni < 4; ni++) {
      int col = nBase + waveN + ni * 16 + (lane & 15);
#pragma unroll
      for (int r = 0; r < 8; r++) {
        int row = mBase + waveM + mi * 16 + r + ((lane >> 4) << 3);
        float val = acc[mi][ni][r];
        if (PERM == 0) {
          Out[(size_t)row * Cc + col] = val;
        } else {
          int bb = row >> 11, mm = row & 2047;   // row = b*2048 + m
          int hh = col >> 6,  dd = col & 63;     // col = h*64 + d
          Out[(((size_t)bb * Hh + hh) * Mm + mm) * Dd + dd] = val;
        }
      }
    }
}

// ---------------------------------------------------------------------------
// Offsets: X[b,n,p] = sum_c q[b,n,c] * offW[c, 2p] + offb[2p]
// ---------------------------------------------------------------------------
__global__ __launch_bounds__(256)
void offsets_kernel(const float* __restrict__ q, const float* __restrict__ offW,
                    const float* __restrict__ offb, float* __restrict__ X) {
  int wave = threadIdx.x >> 5, lane = threadIdx.x & 31;
  int row = blockIdx.x * 8 + wave;            // [0, B*N)
  const float* qrow = q + (size_t)row * Cc;
  float s0 = 0.f, s1 = 0.f, s2 = 0.f, s3 = 0.f;
  for (int c = lane; c < Cc; c += 32) {
    float qv = qrow[c];
    const float* wr = offW + (size_t)c * (2 * Pp);
    s0 += qv * wr[0]; s1 += qv * wr[2]; s2 += qv * wr[4]; s3 += qv * wr[6];
  }
#pragma unroll
  for (int off = 16; off; off >>= 1) {
    s0 += __shfl_xor(s0, off, 32);
    s1 += __shfl_xor(s1, off, 32);
    s2 += __shfl_xor(s2, off, 32);
    s3 += __shfl_xor(s3, off, 32);
  }
  if (lane == 0) {
    float* xr = X + (size_t)row * Pp;
    xr[0] = s0 + offb[0]; xr[1] = s1 + offb[2];
    xr[2] = s2 + offb[4]; xr[3] = s3 + offb[6];
  }
}

// vsumRaw[b,c] = sum_m v[b,m,c]
__global__ __launch_bounds__(256)
void vreduce_kernel(const float* __restrict__ v, float* __restrict__ vsumRaw) {
  int b = blockIdx.x / 3;
  int c = (blockIdx.x % 3) * 256 + threadIdx.x;
  const float* vb = v + (size_t)b * Mm * Cc;
  float acc = 0.f;
  for (int m = 0; m < Mm; m++) acc += vb[(size_t)m * Cc + c];
  vsumRaw[b * Cc + c] = acc;
}

// vsumP[b,co] = sum_c vsumRaw[b,c] * vW[c,co]
__global__ __launch_bounds__(256)
void vproj_kernel(const float* __restrict__ vsumRaw, const float* __restrict__ vW,
                  float* __restrict__ vsumP) {
  int b = blockIdx.x / 3;
  int co = (blockIdx.x % 3) * 256 + threadIdx.x;
  float acc = 0.f;
  for (int c = 0; c < Cc; c++) acc += vsumRaw[b * Cc + c] * vW[(size_t)c * Cc + co];
  vsumP[b * Cc + co] = acc;
}

// Wv[b,h,j] = sum_d vsumP[b, h*64+d] * projW[h*64+d, j]
__global__ __launch_bounds__(256)
void wv_kernel(const float* __restrict__ vsumP, const float* __restrict__ projW,
               float* __restrict__ Wv) {
  int idx = blockIdx.x * 256 + threadIdx.x;   // [0, B*H*768)
  int j = idx % Cc;
  int bh = idx / Cc;
  int h = bh % Hh, b = bh / Hh;
  float acc = 0.f;
  for (int d = 0; d < Dd; d++)
    acc += vsumP[b * Cc + h * Dd + d] * projW[(size_t)(h * Dd + d) * Cc + j];
  Wv[idx] = acc;
}

// ---------------------------------------------------------------------------
// Sample + dot: S[b,h,n]. One block per (b,n); 12 waves, wave = head.
// KP (50MB) is L2-resident so the strided gathers hit L2.
// ---------------------------------------------------------------------------
__global__ __launch_bounds__(384)
void sample_dot_kernel(const float* __restrict__ QP, const float* __restrict__ KP,
                       const float* __restrict__ X, float* __restrict__ S) {
  int bn = blockIdx.x;                 // b*2048 + n
  int b = bn >> 11, n = bn & 2047;
  int h = threadIdx.x >> 5, lane = threadIdx.x & 31;
  const float* qrow = QP + (size_t)bn * Cc + h * Dd;
  float q0 = qrow[lane], q1 = qrow[lane + 32];
  const float* img = KP + (size_t)(b * Hh + h) * (Dd * Mm);  // flat [Dh][M]
  const float* xr = X + (size_t)bn * Pp;
  float partial = 0.f;
#pragma unroll
  for (int p = 0; p < Pp; p++) {
    float xv = xr[p];
    float xf = floorf(xv);
    int   i0 = (int)xf;
    float w  = xv - xf;
    int   i1 = i0 + 1;
    float w0 = (i0 >= 0 && i0 < Mm) ? (1.f - w) : 0.f;
    float w1 = (i1 >= 0 && i1 < Mm) ? w : 0.f;
    int c0 = min(max(i0, 0), Mm - 1);
    int c1 = min(max(i1, 0), Mm - 1);
    float g0a = img[(size_t)lane * Mm + c0];
    float g0b = img[(size_t)(lane + 32) * Mm + c0];
    float g1a = img[(size_t)lane * Mm + c1];
    float g1b = img[(size_t)(lane + 32) * Mm + c1];
    partial += q0 * (w0 * g0a + w1 * g1a) + q1 * (w0 * g0b + w1 * g1b);
  }
#pragma unroll
  for (int off = 16; off; off >>= 1) partial += __shfl_xor(partial, off, 32);
  if (lane == 0)
    S[(size_t)(b * Hh + h) * Nn + n] = partial * (0.125f * 0.25f); // Dh^-0.5 / P
}

// ---------------------------------------------------------------------------
// Softmax over n for each (b,h) row of 2048.
// ---------------------------------------------------------------------------
__global__ __launch_bounds__(256)
void softmax_kernel(const float* __restrict__ S, float* __restrict__ A) {
  __shared__ float red[256];
  int row = blockIdx.x, tid = threadIdx.x;
  const float* s = S + (size_t)row * Nn;
  float* a = A + (size_t)row * Nn;
  float mx = -3.4e38f;
  for (int i = tid; i < Nn; i += 256) mx = fmaxf(mx, s[i]);
  red[tid] = mx; __syncthreads();
  for (int st = 128; st; st >>= 1) {
    if (tid < st) red[tid] = fmaxf(red[tid], red[tid + st]);
    __syncthreads();
  }
  mx = red[0]; __syncthreads();
  float sum = 0.f;
  for (int i = tid; i < Nn; i += 256) { float e = expf(s[i] - mx); a[i] = e; sum += e; }
  red[tid] = sum; __syncthreads();
  for (int st = 128; st; st >>= 1) {
    if (tid < st) red[tid] += red[tid + st];
    __syncthreads();
  }
  float inv = 1.f / red[0];
  for (int i = tid; i < Nn; i += 256) a[i] *= inv;
}

// ---------------------------------------------------------------------------
// Final: out[b,n,j] = sum_h attn[b,h,n] * Wv[b,h,j] + projb[j]
// Wv[b] (36 KB contiguous) is DMA'd into LDS by the Tensor Data Mover:
// D# per ISA 8.3/8.4 — count=1, type=2(image), data_size=2(4B),
// tensor_dim0 = tile_dim0 = 9216 (one-row tile), wait via s_wait_tensorcnt.
// ---------------------------------------------------------------------------
__global__ __launch_bounds__(256)
void final_kernel(const float* __restrict__ Att, const float* __restrict__ Wv,
                  const float* __restrict__ projb, float* __restrict__ Out) {
  __shared__ float sWv[Hh * Cc];   // 36 KB
  __shared__ float sa[16][Hh];
  int b = blockIdx.x >> 7;         // 128 n-chunks per batch
  int nc = blockIdx.x & 127;
  int tid = threadIdx.x;

  if ((tid >> 5) == 0) {           // wave 0 issues the TDM (EXEC ignored)
    unsigned long long ga = (unsigned long long)(uintptr_t)(Wv + (size_t)b * Hh * Cc);
    unsigned ldsOff = (unsigned)(uintptr_t)(&sWv[0]);     // LDS byte offset (flat low 32)
    v4u g0 = { 0x1u,                                      // count=1, user descriptor
               ldsOff,                                    // lds_addr
               (unsigned)(ga & 0xFFFFFFFFu),              // global_addr[31:0]
               (unsigned)(((ga >> 32) & 0x1FFFFFFu) | (2u << 30)) }; // ga[56:32] | type=2
    v8i_ g1 = { (int)0x00020000,                          // data_size=2 (4 bytes)
                (int)(((Hh * Cc) & 0xFFFF) << 16),        // tensor_dim0[15:0] @ bits 63:48
                (int)(((Hh * Cc) >> 16) << 0) | (1 << 16),// tensor_dim0[31:16], tensor_dim1=1
                (int)(((Hh * Cc) & 0xFFFF) << 16),        // tile_dim0 @ bits 127:112
                1,                                        // tile_dim1 = 1
                Hh * Cc,                                  // tensor_dim0_stride[31:0]
                0, 0 };
    v4i_ gz4 = { 0, 0, 0, 0 };
    v8i_ gz8 = { 0, 0, 0, 0, 0, 0, 0, 0 };
    __builtin_amdgcn_tensor_load_to_lds(g0, g1, gz4, gz4, gz8, 0);
    __builtin_amdgcn_s_wait_tensorcnt(0);
  }
  if (tid < 16 * Hh) {
    int h = tid % Hh, nn = tid / Hh;
    sa[nn][h] = Att[(size_t)(b * Hh + h) * Nn + nc * 16 + nn];
  }
  __syncthreads();

  for (int nn = 0; nn < 16; nn++) {
    int n = nc * 16 + nn;
    for (int j = tid; j < Cc; j += 256) {
      float acc = projb[j];
#pragma unroll
      for (int h = 0; h < Hh; h++) acc += sa[nn][h] * sWv[h * Cc + j];
      Out[(size_t)(b * Nn + n) * Cc + j] = acc;
    }
  }
}

// ---------------------------------------------------------------------------
extern "C" void kernel_launch(void* const* d_in, const int* in_sizes, int n_in,
                              void* d_out, int out_size, void* d_ws, size_t ws_size,
                              hipStream_t stream) {
  (void)in_sizes; (void)n_in; (void)out_size; (void)ws_size;
  const float* q     = (const float*)d_in[0];
  const float* k     = (const float*)d_in[1];
  const float* v     = (const float*)d_in[2];
  const float* qW    = (const float*)d_in[3];
  const float* kW    = (const float*)d_in[4];
  const float* vW    = (const float*)d_in[5];
  const float* offW  = (const float*)d_in[6];
  const float* offb  = (const float*)d_in[7];
  const float* projW = (const float*)d_in[8];
  const float* projb = (const float*)d_in[9];
  float* out = (float*)d_out;

  float* ws  = (float*)d_ws;
  float* QP  = ws;                              // B*N*C   = 12,582,912 f32
  float* KP  = QP  + (size_t)Bb * Nn * Cc;      // B*M*C   (stored as [B,H,M,Dh])
  float* X   = KP  + (size_t)Bb * Mm * Cc;      // B*N*P
  float* S   = X   + (size_t)Bb * Nn * Pp;      // B*H*N
  float* ATT = S   + (size_t)Bb * Hh * Nn;      // B*H*N
  float* VR  = ATT + (size_t)Bb * Hh * Nn;      // B*C
  float* VP  = VR  + (size_t)Bb * Cc;           // B*C
  float* WV  = VP  + (size_t)Bb * Cc;           // B*H*C
  float* fend = WV + (size_t)Bb * Hh * Cc;

  __bf16* AFq = (__bf16*)fend;                  // 12,582,912 bf16
  __bf16* AFk = AFq + (size_t)Bb * Nn * Cc;     // 12,582,912 bf16
  __bf16* WFq = AFk + (size_t)Bb * Mm * Cc;     // 589,824 bf16
  __bf16* WFk = WFq + (size_t)Cc * Cc;          // 589,824 bf16

  const int nElemA = Bb * Nn * Cc;              // 12,582,912
  const int nElemW = Cc * Cc;                   // 589,824

  prep_a_frag_kernel<<<nElemA / 256, 256, 0, stream>>>(q, AFq);
  prep_a_frag_kernel<<<nElemA / 256, 256, 0, stream>>>(k, AFk);
  prep_w_frag_kernel<<<nElemW / 256, 256, 0, stream>>>(qW, WFq);
  prep_w_frag_kernel<<<nElemW / 256, 256, 0, stream>>>(kW, WFk);

  dim3 gGemm(128, 6);   // 16384/128 x 768/128
  gemm_frag_kernel<0><<<gGemm, 256, 0, stream>>>(AFq, WFq, QP);
  gemm_frag_kernel<1><<<gGemm, 256, 0, stream>>>(AFk, WFk, KP);

  offsets_kernel  <<<(Bb * Nn) / 8, 256, 0, stream>>>(q, offW, offb, X);
  vreduce_kernel  <<<Bb * 3, 256, 0, stream>>>(v, VR);
  vproj_kernel    <<<Bb * 3, 256, 0, stream>>>(VR, vW, VP);
  wv_kernel       <<<(Bb * Hh * Cc) / 256, 256, 0, stream>>>(VP, projW, WV);
  sample_dot_kernel<<<Bb * Nn, 384, 0, stream>>>(QP, KP, X, S);
  softmax_kernel  <<<Bb * Hh, 256, 0, stream>>>(S, ATT);
  final_kernel    <<<Bb * (Nn / 16), 256, 0, stream>>>(ATT, WV, projb, out);
}
